// TemporalCNNAttention_29618094474195
// MI455X (gfx1250) — matrix operand order repaired and checked by reading, not compiled, not run on hardware
//
#include <hip/hip_runtime.h>
#include <math.h>

#define E_DIM   128
#define W_DIM   16
#define P_DIM   4096
#define B_DIM   8
#define PT      32
#define EP      64                // E_DIM/2 packed bf16 pairs
#define EPP     68                // padded row stride (+16B) to break bank aliasing
#define NPB     (P_DIM / PT)      // 128 p-tiles
#define THREADS 256
#define NGROUPS 4

typedef __attribute__((ext_vector_type(16))) __bf16 v16bf;
typedef __attribute__((ext_vector_type(2)))  __bf16 v2bf;
typedef __attribute__((ext_vector_type(2)))  float  v2f;
typedef __attribute__((ext_vector_type(8)))  float  v8f;

union Frag { v16bf v; unsigned int u[8]; uint4 q[2]; };

// pack two floats into a bf16 pair {y.hi, x.lo} -> single v_cvt_pk_bf16_f32
__device__ __forceinline__ unsigned int pk_bf16(float x, float y) {
  v2f f = {x, y};
  v2bf h = __builtin_convertvector(f, v2bf);
  union { v2bf h; unsigned int u; } r; r.h = h;
  return r.u;
}
__device__ __forceinline__ float bflo(unsigned int u) { return __uint_as_float(u << 16); }
__device__ __forceinline__ float bfhi(unsigned int u) { return __uint_as_float(u & 0xFFFF0000u); }

__device__ __forceinline__ float gelu_exact(float x) {
  return 0.5f * x * (1.0f + erff(x * 0.70710678118654752f));
}

// ---- fragment loaders: 2 aligned b128 LDS reads each, land directly in place ----
// A (16x32 bf16), weights in LDS as [m][EPP] packed pairs
__device__ __forceinline__ void load_A_fragP(const unsigned int* __restrict__ w,
                                             int m0, int k0, int lane, Frag& a) {
  const int m = m0 + (lane & 15);
  const int h = lane >> 4;
  const unsigned int* base = w + m * EPP + (k0 >> 1);
  a.q[0] = *(const uint4*)(base + 4 * h);        // K = 8h + 0..7
  a.q[1] = *(const uint4*)(base + 8 + 4 * h);    // K = 16 + 8h + 0..7
}
// B (32x16 bf16), operand in LDS transposed as [n][EPP] packed pairs
__device__ __forceinline__ void load_B_fragT(const unsigned int* __restrict__ src,
                                             int k0, int n0, int lane, Frag& b) {
  const unsigned int* row = src + (n0 + (lane & 15)) * EPP + (k0 >> 1) + ((lane >> 4) << 2);
  b.q[0] = *(const uint4*)(row);
  b.q[1] = *(const uint4*)(row + 8);
}

// stage 128x128 f32 weights -> packed bf16 pairs dst[m][kp] (optionally A = srcT)
__device__ __forceinline__ void stage_wP(unsigned int* __restrict__ dst,
                                         const float* __restrict__ src,
                                         int tid, bool transpose) {
  for (int i = tid; i < E_DIM * EP; i += THREADS) {
    int m = i >> 6, kp = i & 63;
    float x, y;
    if (!transpose) { x = src[m * E_DIM + 2 * kp];     y = src[m * E_DIM + 2 * kp + 1]; }
    else            { x = src[(2 * kp) * E_DIM + m];   y = src[(2 * kp + 1) * E_DIM + m]; }
    dst[m * EPP + kp] = pk_bf16(x, y);
  }
}

// D = W(128x128 packed) x B(128xPT packed,[n][kp]) [+bias]; compile-time outputs
template <bool HAS_BIAS, bool WRITE_P, bool WRITE_32>
__device__ __forceinline__ void gemm128P(const unsigned int* __restrict__ wlds,
                                         const unsigned int* __restrict__ Bsrc,
                                         unsigned int* __restrict__ DstP,   // [n][kp]
                                         float* __restrict__ Dst32,         // [f][n]
                                         const float* __restrict__ biasL,   // LDS
                                         int wave, int lane) {
  const int m0 = wave * 16;
  const int n  = lane & 15;
  const int h  = lane >> 4;
#pragma unroll
  for (int n0 = 0; n0 < PT; n0 += 16) {
    v8f c = {0.f, 0.f, 0.f, 0.f, 0.f, 0.f, 0.f, 0.f};
#pragma unroll
    for (int k0 = 0; k0 < E_DIM; k0 += 32) {
      Frag a, bf;
      load_A_fragP(wlds, m0, k0, lane, a);
      load_B_fragT(Bsrc, k0, n0, lane, bf);
      c = __builtin_amdgcn_wmma_f32_16x16x32_bf16(false, a.v, false, bf.v,
                                                  (short)0, c, false, false);
    }
    const int nn = n0 + n;
    float vv[8];
#pragma unroll
    for (int r = 0; r < 8; ++r) {
      vv[r] = c[r];
      if (HAS_BIAS) vv[r] += biasL[m0 + r + 8 * h];
    }
    if (WRITE_32) {
#pragma unroll
      for (int r = 0; r < 8; ++r) Dst32[(m0 + r + 8 * h) * PT + nn] = vv[r];
    }
    if (WRITE_P) {                                // one b128 store per n-tile
      uint4 st;
      st.x = pk_bf16(vv[0], vv[1]);
      st.y = pk_bf16(vv[2], vv[3]);
      st.z = pk_bf16(vv[4], vv[5]);
      st.w = pk_bf16(vv[6], vv[7]);
      *(uint4*)(DstP + nn * EPP + (m0 >> 1) + 4 * h) = st;
    }
  }
}

// LDS: floats 15456*4 = 61824 B ; u32 50048*4 = 200192 B ; total 262016 B (<320KB)
#define SMEM_BYTES (15456u * 4u + 50048u * 4u)

__global__ __launch_bounds__(THREADS, 1)
void fused_attn_kernel(const float* __restrict__ u,
                       const float* __restrict__ emb_w, const float* __restrict__ emb_b,
                       const float* __restrict__ wq, const float* __restrict__ bq,
                       const float* __restrict__ wk, const float* __restrict__ bk,
                       const float* __restrict__ wv, const float* __restrict__ bv,
                       const float* __restrict__ wo, const float* __restrict__ bo,
                       float* __restrict__ xout, float* __restrict__ partial) {
  extern __shared__ __align__(16) char smem[];
  float* u_lds  = (float*)smem;                 // 16*36          = 576
  float* we0    = u_lds + 16 * 36;              // 128
  float* we1    = we0 + 128;
  float* we2    = we1 + 128;
  float* web    = we2 + 128;
  float* bqL    = web + 128;                    // 4 x 128 staged biases
  float* bkL    = bqL + 128;
  float* bvL    = bkL + 128;
  float* boL    = bvL + 128;
  float* lastf  = boL + 128;                    // 128*PT (fp32 residual)
  float* Qf     = lastf + 128 * PT;             // 128*PT (for qb dot)
  float* qkf    = Qf + 128 * PT;                // 128*PT (scores operand)
  float* scores = qkf + 128 * PT;               // 16*PT
  float* attn   = scores + 16 * PT;             // 16*PT
  float* qb     = attn + 16 * PT;               // PT
  float* psum   = qb + PT;                      // 256
  float* psq    = psum + 256;                   // 256
  unsigned int* featP = (unsigned int*)(psq + 256);   // [16][PT][EPP] = 34816
  unsigned int* QP    = featP + 16 * PT * EPP;        // [PT][EPP]     = 2176
  unsigned int* cfP   = QP + PT * EPP;                // 2176
  unsigned int* ctxP  = cfP + PT * EPP;               // 2176
  unsigned int* wstg  = ctxP + PT * EPP;              // [128][EPP]    = 8704

  const int tid  = threadIdx.x;
  const int lane = tid & 31, wave = tid >> 5;
  const int pblk = blockIdx.x, b = blockIdx.y;
  const int p0   = pblk * PT;

  // ---- phase 0: stage u window (zero conv padding) + conv weights + biases ----
  for (int i = tid; i < W_DIM * 34; i += THREADS) {
    int w = i / 34, j = i % 34;
    int pg = p0 - 1 + j;
    float v = 0.f;
    if (pg >= 0 && pg < P_DIM) v = u[((size_t)b * W_DIM + w) * P_DIM + pg];
    u_lds[w * 36 + j] = v;
  }
  if (tid < 128) {
    we0[tid] = emb_w[tid * 3 + 0];
    we1[tid] = emb_w[tid * 3 + 1];
    we2[tid] = emb_w[tid * 3 + 2];
    web[tid] = emb_b[tid];
    bqL[tid] = bq[tid];
    bkL[tid] = bk[tid];
    bvL[tid] = bv[tid];
    boL[tid] = bo[tid];
  }
  __builtin_prefetch(wq + tid, 0, 0);           // global_prefetch_b8: warm wq
  __syncthreads();

  // ---- phase 1: conv(k=3) + exact GELU -> packed bf16 features [w][p][ep] ----
  for (int i = tid; i < W_DIM * PT * EP; i += THREADS) {
    int w  = i >> 11;            // PT*EP = 2048
    int p  = (i >> 6) & 31;
    int ep = i & 63;
    const float* ur = u_lds + w * 36 + p;
    int ea = 2 * ep, eb2 = 2 * ep + 1;
    float x0 = gelu_exact(we0[ea]  * ur[0] + we1[ea]  * ur[1] + we2[ea]  * ur[2] + web[ea]);
    float x1 = gelu_exact(we0[eb2] * ur[0] + we1[eb2] * ur[1] + we2[eb2] * ur[2] + web[eb2]);
    featP[(w * PT + p) * EPP + ep] = pk_bf16(x0, x1);
    if (w == W_DIM - 1) { lastf[ea * PT + p] = x0; lastf[eb2 * PT + p] = x1; }
  }
  __syncthreads();

  // ---- phase 2: Q = wq @ last + bq  (B operand = packed feat slab at w=15) ----
  stage_wP(wstg, wq, tid, false);
  __syncthreads();
  gemm128P<true, true, true>(wstg, featP + (W_DIM - 1) * PT * EPP, QP, Qf, bqL, wave, lane);
  __syncthreads();

  // ---- phase 3: qk = wk^T @ Q ; qb[p] = Q[:,p] . bk ----
  stage_wP(wstg, wk, tid, true);
  if (tid < PT) {
    float s = 0.f;
    for (int f = 0; f < E_DIM; ++f) s += Qf[f * PT + tid] * bkL[f];
    qb[tid] = s;
  }
  __syncthreads();
  gemm128P<false, false, true>(wstg, QP, nullptr, qkf, nullptr, wave, lane);
  __syncthreads();

  // ---- phase 4: scores + softmax over W=16 ----
  const float rsqrtE = 0.08838834764831845f;    // 1/sqrt(128)
  for (int i = tid; i < W_DIM * PT; i += THREADS) {
    int w = i >> 5, p = i & 31;
    const unsigned int* fr = featP + (w * PT + p) * EPP;   // contiguous 64 pairs
    float s = 0.f;
    for (int ep = 0; ep < EP; ++ep) {
      unsigned int pk = fr[ep];
      s += qkf[(2 * ep) * PT + p] * bflo(pk) + qkf[(2 * ep + 1) * PT + p] * bfhi(pk);
    }
    scores[w * PT + p] = (s + qb[p]) * rsqrtE;
  }
  __syncthreads();
  if (tid < PT) {
    float m = -1e30f;
    for (int w = 0; w < W_DIM; ++w) m = fmaxf(m, scores[w * PT + tid]);
    float ex[W_DIM]; float s = 0.f;
    for (int w = 0; w < W_DIM; ++w) { ex[w] = __expf(scores[w * PT + tid] - m); s += ex[w]; }
    float inv = 1.f / s;
    for (int w = 0; w < W_DIM; ++w) attn[w * PT + tid] = ex[w] * inv;
  }
  __syncthreads();

  // ---- phase 5: cf[p][ep] = sum_w attn * feat ; stage wv ----
  for (int i = tid; i < PT * EP; i += THREADS) {
    int p  = i >> 6;
    int ep = i & 63;
    float s0 = 0.f, s1 = 0.f;
    for (int w = 0; w < W_DIM; ++w) {
      unsigned int pk = featP[(w * PT + p) * EPP + ep];
      float a = attn[w * PT + p];
      s0 += a * bflo(pk);
      s1 += a * bfhi(pk);
    }
    cfP[p * EPP + ep] = pk_bf16(s0, s1);
  }
  stage_wP(wstg, wv, tid, false);
  __syncthreads();

  // ---- phase 6: context = wv @ cf + bv  (sum_w attn == 1 absorbs bv) ----
  gemm128P<true, true, false>(wstg, cfP, ctxP, nullptr, bvL, wave, lane);
  __syncthreads();
  stage_wP(wstg, wo, tid, false);
  __syncthreads();

  // ---- phase 7: x = wo @ context + bo + last ; write out + GN partials ----
  {
    const int m0 = wave * 16;
    const int n  = lane & 15;
    const int h  = lane >> 4;
    float s1 = 0.f, s2 = 0.f;
#pragma unroll
    for (int n0 = 0; n0 < PT; n0 += 16) {
      v8f c = {0.f, 0.f, 0.f, 0.f, 0.f, 0.f, 0.f, 0.f};
#pragma unroll
      for (int k0 = 0; k0 < E_DIM; k0 += 32) {
        Frag a, bf;
        load_A_fragP(wstg, m0, k0, lane, a);
        load_B_fragT(ctxP, k0, n0, lane, bf);
        c = __builtin_amdgcn_wmma_f32_16x16x32_bf16(false, a.v, false, bf.v,
                                                    (short)0, c, false, false);
      }
#pragma unroll
      for (int r = 0; r < 8; ++r) {
        int f = m0 + r + 8 * h;                 // whole tile lies in group wave>>1
        float x = c[r] + boL[f] + lastf[f * PT + n0 + n];
        xout[(((size_t)b * E_DIM + f) << 12) + p0 + n0 + n] = x;
        s1 += x; s2 += x * x;
      }
    }
    psum[tid] = s1; psq[tid] = s2;
  }
  __syncthreads();
  if (tid < NGROUPS) {
    int g = tid;
    float s1 = 0.f, s2 = 0.f;
    for (int t = g * 64; t < g * 64 + 64; ++t) { s1 += psum[t]; s2 += psq[t]; }
    int idx = ((b * NGROUPS + g) * NPB + pblk) * 2;
    partial[idx] = s1; partial[idx + 1] = s2;
  }
}

// deterministic reduce of per-block partials -> (mu, rsig) per (b, group)
__global__ void gn_stats_kernel(const float* __restrict__ partial, float* __restrict__ stats) {
  int t = threadIdx.x;
  if (t < B_DIM * NGROUPS) {
    float s1 = 0.f, s2 = 0.f;
    for (int j = 0; j < NPB; ++j) {
      s1 += partial[(t * NPB + j) * 2];
      s2 += partial[(t * NPB + j) * 2 + 1];
    }
    const float invN = 1.f / (float)((E_DIM / NGROUPS) * P_DIM);
    float mu  = s1 * invN;
    float var = s2 * invN - mu * mu;
    stats[t * 2]     = mu;
    stats[t * 2 + 1] = rsqrtf(var + 1e-5f);
  }
}

__global__ __launch_bounds__(256)
void gn_apply_kernel(float4* __restrict__ x, const float* __restrict__ stats,
                     const float* __restrict__ gw, const float* __restrict__ gb) {
  int i4 = blockIdx.x * 256 + threadIdx.x;      // 1M float4s total
  int b  = i4 >> 17;                            // E*P/4 = 131072 per batch
  int e  = (i4 >> 10) & 127;                    // P/4   = 1024 per channel
  int g  = e >> 5;
  float mu = stats[(b * NGROUPS + g) * 2];
  float rs = stats[(b * NGROUPS + g) * 2 + 1];
  float sc = rs * gw[e];
  float sh = gb[e] - mu * sc;                   // y = x*sc + sh
  float4 v = x[i4];
  v.x = v.x * sc + sh; v.y = v.y * sc + sh;
  v.z = v.z * sc + sh; v.w = v.w * sc + sh;
  x[i4] = v;
}

extern "C" void kernel_launch(void* const* d_in, const int* in_sizes, int n_in,
                              void* d_out, int out_size, void* d_ws, size_t ws_size,
                              hipStream_t stream) {
  (void)in_sizes; (void)n_in; (void)out_size; (void)ws_size;
  const float* u    = (const float*)d_in[0];
  const float* embw = (const float*)d_in[1];
  const float* embb = (const float*)d_in[2];
  const float* wq   = (const float*)d_in[3];
  const float* bq   = (const float*)d_in[4];
  const float* wk   = (const float*)d_in[5];
  const float* bk   = (const float*)d_in[6];
  const float* wv   = (const float*)d_in[7];
  const float* bv   = (const float*)d_in[8];
  const float* wo   = (const float*)d_in[9];
  const float* bo   = (const float*)d_in[10];
  const float* gw   = (const float*)d_in[11];
  const float* gb   = (const float*)d_in[12];

  float* x       = (float*)d_out;
  float* partial = (float*)d_ws;                           // 8*4*128*2 floats
  float* stats   = partial + B_DIM * NGROUPS * NPB * 2;    // 64 floats

  dim3 grid(NPB, B_DIM);
  fused_attn_kernel<<<grid, THREADS, SMEM_BYTES, stream>>>(
      u, embw, embb, wq, bq, wk, bk, wv, bv, wo, bo, x, partial);
  gn_stats_kernel<<<1, 32, 0, stream>>>(partial, stats);
  gn_apply_kernel<<<(B_DIM * E_DIM * P_DIM) / 4 / 256, 256, 0, stream>>>(
      (float4*)x, stats, gw, gb);
}
